// FlexCPB2DIdxSentinel_55121610277139
// MI455X (gfx1250) — matrix operand order repaired and checked by reading, not compiled, not run on hardware
//
#include <hip/hip_runtime.h>

// Problem constants (from reference)
#define B_   4
#define N_   1028
#define R_   4
#define S_   32
#define H_   12
#define HID_ 32
#define D_   64
#define L_   3969          // (2S-1)^2
#define LP1  (L_ + 1)

#define KCH    32
#define NWAVE  8
#define QTILE  (NWAVE * 16)   // 128 q rows per block
#define VSTR   40             // padded bf16 row stride for V^T tile (80B, 16B aligned)
#define NFULL  (N_ / KCH)     // 32 full chunks; tail has 4 valid keys

#define LOG2E 1.44269504088896340736f

typedef __attribute__((ext_vector_type(16))) __bf16 v16bf;
typedef __attribute__((ext_vector_type(2)))  __bf16 v2bf;
typedef __attribute__((ext_vector_type(8)))  float  v8f;

// Two-operand packed f32->bf16 convert (RTNE). The clang builtin is absent on this
// toolchain but the instruction exists (backend emits it for scalar fptrunc), so
// use inline asm to get one VALU per pair instead of cvt+shift+or merge chains.
static __device__ __forceinline__ unsigned pack_bf2(float lo, float hi) {
#if __has_builtin(__builtin_amdgcn_cvt_pk_bf16_f32)
    return __builtin_bit_cast(unsigned, __builtin_amdgcn_cvt_pk_bf16_f32(lo, hi));
#else
    unsigned r;
    asm("v_cvt_pk_bf16_f32 %0, %1, %2" : "=v"(r) : "v"(lo), "v"(hi));
    return r;
#endif
}

// Raw exp2 (v_exp_f32 is natively base-2)
static __device__ __forceinline__ float fast_exp2(float x) {
#if __has_builtin(__builtin_amdgcn_exp2f)
    return __builtin_amdgcn_exp2f(x);
#else
    return exp2f(x);
#endif
}

// DPP lane-permute (VALU, no LDS) for 16-lane butterfly reductions
template<int CTRL>
static __device__ __forceinline__ float dppf(float x) {
    int xi = __builtin_bit_cast(int, x);
    return __builtin_bit_cast(float,
        __builtin_amdgcn_update_dpp(xi, xi, CTRL, 0xF, 0xF, true));
}
static __device__ __forceinline__ float rowmax16(float x) {
    x = fmaxf(x, dppf<0xB1>(x));    // quad_perm [1,0,3,2]  (xor 1)
    x = fmaxf(x, dppf<0x4E>(x));    // quad_perm [2,3,0,1]  (xor 2)
    x = fmaxf(x, dppf<0x141>(x));   // row_half_mirror      (groups of 4 -> 8)
    x = fmaxf(x, dppf<0x140>(x));   // row_mirror           (groups of 8 -> 16)
    return x;
}

// CPB MLP table, stored TRANSPOSED: bt_t[h][0]=0 sentinel, bt_t[h][l+1]
// (cold one-shot kernel: keep loops rolled to minimize static footprint)
__global__ void FlexCPB_table_kernel(const float* __restrict__ rel_table,
                                     const float* __restrict__ W1,
                                     const float* __restrict__ b1,
                                     const float* __restrict__ W2,
                                     float* __restrict__ bt_t) {
    int t = blockIdx.x * blockDim.x + threadIdx.x;
    if (t > L_) return;
    if (t == 0) {
#pragma clang loop unroll(disable)
        for (int h = 0; h < H_; h++) bt_t[(size_t)h * LP1] = 0.0f;
        return;
    }
    int l = t - 1;
    float r0 = rel_table[l * 2 + 0], r1 = rel_table[l * 2 + 1];
    float hid[HID_];
#pragma clang loop unroll(disable)
    for (int j = 0; j < HID_; j++) {
        float x = W1[j * 2 + 0] * r0 + W1[j * 2 + 1] * r1 + b1[j];
        hid[j] = 0.5f * x * (1.0f + erff(x * 0.70710678118654752f)); // exact GELU
    }
#pragma clang loop unroll(disable)
    for (int h = 0; h < H_; h++) {
        float acc = 0.0f;
#pragma clang loop unroll(disable)
        for (int j = 0; j < HID_; j++) acc += W2[h * HID_ + j] * hid[j];
        bt_t[(size_t)h * LP1 + t] = acc;
    }
}

// gmu = LOG2E * sigmoid(gamma[h]) * sign(mu)*log1p(|mu|)  (log2-domain softmax)
__global__ void FlexCPB_mu_kernel(const float* __restrict__ mu,
                                  const float* __restrict__ gamma,
                                  float* __restrict__ gmuq,
                                  float* __restrict__ gmuk) {
    int t = blockIdx.x * blockDim.x + threadIdx.x;
    const int total = B_ * H_ * N_;
    if (t >= total) return;
    int b = t / (H_ * N_);
    int rem = t - b * (H_ * N_);
    int h = rem / N_;
    int n = rem - h * N_;
    float gam = LOG2E / (1.0f + __expf(-gamma[h]));
    float mq = mu[((size_t)b * 2 * H_ + h) * N_ + n];
    float mk = mu[((size_t)b * 2 * H_ + H_ + h) * N_ + n];
    gmuq[t] = gam * copysignf(log1pf(fabsf(mq)), mq);
    gmuk[t] = gam * copysignf(log1pf(fabsf(mk)), mk);
}

__global__ __launch_bounds__(256) void FlexCPB_attn_kernel(
        const float* __restrict__ q, const float* __restrict__ k, const float* __restrict__ v,
        const float* __restrict__ bt_t,
        const float* __restrict__ gmuq, const float* __restrict__ gmuk,
        float* __restrict__ out) {
    __shared__ unsigned short Klds[2][KCH * D_];       // double-buffered [32 keys][64 d] bf16
    __shared__ unsigned short Vlds[2][D_ * VSTR];      // double-buffered [64 d][40] bf16 (V^T)
    __shared__ unsigned short Plds[NWAVE * 16 * KCH];  // per-wave 16x32 P bf16

    const int qt = blockIdx.x, h = blockIdx.y, b = blockIdx.z;
    const int tid  = threadIdx.x;
    const int wave = tid >> 5, lane = tid & 31, ln16 = lane & 15, half = lane >> 4;

    const size_t bh = (size_t)(b * H_ + h);
    const float* qg   = q    + bh * N_ * D_;
    const float* kg   = k    + bh * N_ * D_;
    const float* vg   = v    + bh * N_ * D_;
    const float* muqh = gmuq + bh * N_;
    const float* mukh = gmuk + bh * N_;
    const float* bth  = bt_t + (size_t)h * LP1;

    const int qbase = qt * QTILE + wave * 16;

    union BF16Frag { unsigned u[8]; v16bf v; uint4 qv[2]; };

    // ---- Q A-frags (ISA A layout), pre-scaled by LOG2E/sqrt(D) (log2-domain scores)
    const float qscale = 0.125f * LOG2E;
    int qrow = qbase + ln16; if (qrow >= N_) qrow = N_ - 1;
    const float* qp = qg + (size_t)qrow * D_;
    BF16Frag qa[2];
#pragma unroll
    for (int f = 0; f < 2; f++) {
        int d0 = f * 32 + 8 * half;
#pragma unroll
        for (int e = 0; e < 4; e++)
            qa[f].u[e]     = pack_bf2(qp[d0 + 2*e] * qscale,      qp[d0 + 2*e + 1] * qscale);
#pragma unroll
        for (int e = 0; e < 4; e++)
            qa[f].u[4 + e] = pack_bf2(qp[d0 + 16 + 2*e] * qscale, qp[d0 + 16 + 2*e + 1] * qscale);
    }

    // all-ones B frag: P @ ones accumulates row-sums of P
    BF16Frag onesb;
#pragma unroll
    for (int e = 0; e < 8; e++) onesb.u[e] = 0x3F803F80u;

    // per-slot row state (C layout: slot r -> row i = qbase + r + 8*half)
    float mur[8]; int rowcode[8]; bool rowok[8];
#pragma unroll
    for (int r = 0; r < 8; r++) {
        int i = qbase + r + 8 * half;
        int ic = i < N_ ? i : N_ - 1;
        mur[r]  = muqh[ic];
        rowok[r] = (i >= R_) && (i < N_);
        int a = i - R_;
        if (a < 0) a = 0;
        if (a > N_ - R_ - 1) a = N_ - R_ - 1;
        rowcode[r] = (a >> 5) * 63 + (a & 31);   // Toeplitz row code
    }

    v8f accO[4], accL;
#pragma unroll
    for (int r = 0; r < 8; r++) {
        accO[0][r] = 0.f; accO[1][r] = 0.f; accO[2][r] = 0.f; accO[3][r] = 0.f;
        accL[r] = 0.f;
    }
    float m[8];
#pragma unroll
    for (int r = 0; r < 8; r++) m[r] = -1e30f;

    unsigned short* Pw = Plds + wave * 16 * KCH;

    // cooperative-staging thread mappings (256 threads)
    const int kjl = tid >> 3;             // 0..31 key row
    const int kd0 = (tid & 7) * 8;        // 0..56
    const int vd  = (tid & 31) * 2;       // 0..62 (d pair)
    const int vj  = (tid >> 5) * 4;       // 0..28 (4 keys)

    const float* kptr = kg + kjl * D_ + kd0;
    const float* vptr = vg + vj * D_ + vd;

    // register prefetch state
    float4 kf0, kf1;
    float2 vf0, vf1, vf2, vf3;

    auto load_full = [&]() {
        kf0 = *(const float4*)(kptr);
        kf1 = *(const float4*)(kptr + 4);
        vf0 = *(const float2*)(vptr);
        vf1 = *(const float2*)(vptr + D_);
        vf2 = *(const float2*)(vptr + 2 * D_);
        vf3 = *(const float2*)(vptr + 3 * D_);
        kptr += KCH * D_;
        vptr += KCH * D_;
    };
    auto load_tail = [&]() {
        int jg = NFULL * KCH + kjl; if (jg >= N_) jg = N_ - 1;
        const float* p = kg + (size_t)jg * D_ + kd0;
        kf0 = *(const float4*)(p);
        kf1 = *(const float4*)(p + 4);
        int j0g = NFULL * KCH + vj;
        int ja = j0g     < N_ ? j0g     : N_ - 1;
        int jb = j0g + 1 < N_ ? j0g + 1 : N_ - 1;
        int jc = j0g + 2 < N_ ? j0g + 2 : N_ - 1;
        int jd = j0g + 3 < N_ ? j0g + 3 : N_ - 1;
        vf0 = *(const float2*)(vg + (size_t)ja * D_ + vd);
        vf1 = *(const float2*)(vg + (size_t)jb * D_ + vd);
        vf2 = *(const float2*)(vg + (size_t)jc * D_ + vd);
        vf3 = *(const float2*)(vg + (size_t)jd * D_ + vd);
    };
    auto stage = [&](int buf) {
        uint4 kp;
        kp.x = pack_bf2(kf0.x, kf0.y); kp.y = pack_bf2(kf0.z, kf0.w);
        kp.z = pack_bf2(kf1.x, kf1.y); kp.w = pack_bf2(kf1.z, kf1.w);
        *(uint4*)(&Klds[buf][kjl * D_ + kd0]) = kp;
        uint2 vp0, vp1;
        vp0.x = pack_bf2(vf0.x, vf1.x); vp0.y = pack_bf2(vf2.x, vf3.x);
        vp1.x = pack_bf2(vf0.y, vf1.y); vp1.y = pack_bf2(vf2.y, vf3.y);
        *(uint2*)(&Vlds[buf][vd * VSTR + vj])       = vp0;
        *(uint2*)(&Vlds[buf][(vd + 1) * VSTR + vj]) = vp1;
    };

    auto compute = [&](int buf, int kbase, bool tail) {
        // ---- QK^T: two 16-key tiles, D=64 as two K=32 WMMA steps ----
        v8f sc[2];
#pragma unroll
        for (int kt = 0; kt < 2; kt++) {
            v8f acc = {0.f, 0.f, 0.f, 0.f, 0.f, 0.f, 0.f, 0.f};
#pragma unroll
            for (int ds = 0; ds < 2; ds++) {
                BF16Frag kb;
                const unsigned short* src = &Klds[buf][(kt * 16 + ln16) * D_ + ds * 32 + 8 * half];
                kb.qv[0] = *(const uint4*)(src);
                kb.qv[1] = *(const uint4*)(src + 16);
                acc = __builtin_amdgcn_wmma_f32_16x16x32_bf16(
                        false, qa[ds].v, false, kb.v, (short)0, acc, false, false);
            }
            sc[kt] = acc;
        }

        // ---- bias + mask + online softmax (log2 domain) ----
        int j0 = kbase + ln16, j1 = j0 + 16;
        int jc0 = (!tail || j0 < N_) ? j0 : N_ - 1;
        int jc1 = (!tail || j1 < N_) ? j1 : N_ - 1;
        float gk0 = mukh[jc0], gk1 = mukh[jc1];
        int c0 = j0 - R_, c1 = j1 - R_;
        int cc0 = (c0 >> 5) * 63 + (c0 & 31);
        int cc1 = (c1 >> 5) * 63 + (c1 & 31);
        bool colok0 = (j0 >= R_) && (!tail || j0 < N_);
        bool colok1 = (j1 >= R_) && (!tail || j1 < N_);
#pragma unroll
        for (int r = 0; r < 8; r++) {
            int l0 = rowcode[r] - cc0 + 1985;   // (l+1) table index
            int l1 = rowcode[r] - cc1 + 1985;
            float bt0 = (rowok[r] && colok0) ? bth[l0] : 0.0f;
            float bt1 = (rowok[r] && colok1) ? bth[l1] : 0.0f;
            float s0 = sc[0][r] + (mur[r] + gk0) * bt0;  // all terms pre-scaled by LOG2E
            float s1 = sc[1][r] + (mur[r] + gk1) * bt1;
            if (tail) {
                if (j0 >= N_) s0 = -1e30f;
                if (j1 >= N_) s1 = -1e30f;
            }
            float mx = rowmax16(fmaxf(s0, s1));
            float mn = fmaxf(m[r], mx);
            float alpha = fast_exp2(m[r] - mn);
            m[r] = mn;
            float p0 = fast_exp2(s0 - mn);
            float p1 = fast_exp2(s1 - mn);
            accO[0][r] *= alpha; accO[1][r] *= alpha;
            accO[2][r] *= alpha; accO[3][r] *= alpha;
            accL[r]    *= alpha;
            unsigned pk = pack_bf2(p0, p1);
            int row = r + 8 * half;
            Pw[row * KCH + ln16]      = (unsigned short)(pk & 0xFFFFu);
            Pw[row * KCH + 16 + ln16] = (unsigned short)(pk >> 16);
        }

        // ---- P·V (+ ones column for row-sums); P re-read in A layout (in-order LDS) ----
        BF16Frag pa;
        const unsigned short* pr = &Pw[ln16 * KCH + 8 * half];
        pa.qv[0] = *(const uint4*)(pr);
        pa.qv[1] = *(const uint4*)(pr + 16);
        accL = __builtin_amdgcn_wmma_f32_16x16x32_bf16(
                false, pa.v, false, onesb.v, (short)0, accL, false, false);
#pragma unroll
        for (int dt = 0; dt < 4; dt++) {
            BF16Frag vb;
            const unsigned short* vr = &Vlds[buf][(dt * 16 + ln16) * VSTR + 8 * half];
            vb.qv[0] = *(const uint4*)(vr);
            vb.qv[1] = *(const uint4*)(vr + 16);
            accO[dt] = __builtin_amdgcn_wmma_f32_16x16x32_bf16(
                    false, pa.v, false, vb.v, (short)0, accO[dt], false, false);
        }
    };

    // ---- software-pipelined main loop: prefetch chunk n+1 while computing chunk n ----
    load_full();                       // chunk 0 into registers
    for (int ch = 0; ch < NFULL; ch++) {
        stage(ch & 1);
        __syncthreads();
        if (ch + 1 < NFULL) load_full();
        else                load_tail();
        compute(ch & 1, ch * KCH, false);
    }
    stage(NFULL & 1);
    __syncthreads();
    compute(NFULL & 1, NFULL * KCH, true);

    // ---- normalize by row-sum (accL) and store ----
#pragma unroll
    for (int r = 0; r < 8; r++) {
        int i = qbase + r + 8 * half;
        if (i >= N_) continue;
        float inv = 1.0f / accL[r];
        float* op = out + (bh * N_ + (size_t)i) * D_;
        op[ln16]      = accO[0][r] * inv;
        op[16 + ln16] = accO[1][r] * inv;
        op[32 + ln16] = accO[2][r] * inv;
        op[48 + ln16] = accO[3][r] * inv;
    }
}

extern "C" void kernel_launch(void* const* d_in, const int* in_sizes, int n_in,
                              void* d_out, int out_size, void* d_ws, size_t ws_size,
                              hipStream_t stream) {
    const float* q         = (const float*)d_in[0];
    const float* k         = (const float*)d_in[1];
    const float* v         = (const float*)d_in[2];
    const float* mu        = (const float*)d_in[3];
    const float* rel_table = (const float*)d_in[4];
    const float* W1        = (const float*)d_in[5];
    const float* b1        = (const float*)d_in[6];
    const float* W2        = (const float*)d_in[7];
    const float* gamma     = (const float*)d_in[8];
    // d_in[9] = idx_table: reconstructed arithmetically, unused

    float* ws   = (float*)d_ws;
    float* bt_t = ws;                                    // (H, L+1) transposed
    float* gmuq = bt_t + (size_t)H_ * LP1 + 64;          // +64 pad (guarded OOB-by-value reads)
    float* gmuk = gmuq + (size_t)B_ * H_ * N_;

    FlexCPB_table_kernel<<<(L_ + 1 + 127) / 128, 128, 0, stream>>>(rel_table, W1, b1, W2, bt_t);
    FlexCPB_mu_kernel<<<(B_ * H_ * N_ + 255) / 256, 256, 0, stream>>>(mu, gamma, gmuq, gmuk);

    dim3 grid((N_ + QTILE - 1) / QTILE, H_, B_);
    FlexCPB_attn_kernel<<<grid, 256, 0, stream>>>(q, k, v, bt_t, gmuq, gmuk, (float*)d_out);
}
